// DependencyParsingNetwork_41515153883380
// MI455X (gfx1250) — matrix operand order repaired and checked by reading, not compiled, not run on hardware
//
#include <hip/hip_runtime.h>
#include <hip/hip_bf16.h>
#include <math.h>

typedef __attribute__((ext_vector_type(16))) __bf16 v16bf;
typedef __attribute__((ext_vector_type(8)))  __bf16 v8bf;
typedef __attribute__((ext_vector_type(8)))  float  v8f;
typedef __attribute__((ext_vector_type(4)))  float  v4f;

#define GRU_H 200
#define SEQ_T 512
#define SPAD  224   // 200 padded to multiple of 32

__device__ __forceinline__ float fast_tanh(float x) {
#if __has_builtin(__builtin_amdgcn_tanhf)
  return __builtin_amdgcn_tanhf(x);        // native v_tanh_f32 (1 TRANS op)
#elif __has_builtin(__builtin_amdgcn_tanh_f32)
  return __builtin_amdgcn_tanh_f32(x);
#else
  float e = __expf(2.f * x);               // fallback: v_exp + v_rcp
  return 1.f - 2.f / (e + 1.f);
#endif
}

// ---------------------------------------------------------------------------
// 1) Embedding gather + concat
// ---------------------------------------------------------------------------
__global__ void gather_x_kernel(const int* __restrict__ tok, const int* __restrict__ pos,
                                const float* __restrict__ wemb, const float* __restrict__ pemb,
                                float* __restrict__ x, int T) {
  int idx = blockIdx.x * blockDim.x + threadIdx.x;
  if (idx >= T * 320) return;
  int t = idx / 320, c = idx - t * 320;
  x[idx] = (c < 300) ? wemb[(size_t)tok[t] * 300 + c]
                     : pemb[(size_t)pos[t] * 20 + (c - 300)];
}

// ---------------------------------------------------------------------------
// Pack f32 -> zero-padded bf16 (drows x ldd); padding zero => guard-free GEMM.
// ---------------------------------------------------------------------------
__global__ void pack_bf16(const float* __restrict__ src, int lds, int col_off,
                          int rows, int cols,
                          __bf16* __restrict__ dst, int ldd, int drows) {
  int idx = blockIdx.x * blockDim.x + threadIdx.x;
  if (idx >= drows * ldd) return;
  int r = idx / ldd, c = idx - r * ldd;
  float v = (r < rows && c < cols) ? src[(size_t)r * lds + col_off + c] : 0.f;
  dst[idx] = (__bf16)v;
}

__global__ void fill_zero_f32(float* __restrict__ p, int n) {
  int i = blockIdx.x * blockDim.x + threadIdx.x;
  if (i < n) p[i] = 0.f;
}

__global__ void pack_pad_f32(const float* __restrict__ src, int n,
                             float* __restrict__ dst, int np) {
  int i = blockIdx.x * blockDim.x + threadIdx.x;
  if (i < np) dst[i] = (i < n) ? src[i] : 0.f;
}

// ---------------------------------------------------------------------------
// 2) WMMA GEMM on packed bf16: Out[m][ocol+n] = sum_k A[m][k]*B[n][k] (+bias)
//    One wave = 16x32 output tile: A loaded once per K-chunk, feeds TWO
//    independent WMMA accumulation chains (hides XDL RAW latency).
//    B buffer must have rows padded to a multiple of 32.
// ---------------------------------------------------------------------------
__global__ void wmma_gemm_bf16(const __bf16* __restrict__ A, int lda,
                               const __bf16* __restrict__ B, int ldb,
                               const float* __restrict__ bias,
                               float* __restrict__ Out, int ldo, int ocol,
                               int M, int N, int K) {
  const int lane = threadIdx.x & 31;
  const int wave = blockIdx.x * (blockDim.x >> 5) + (threadIdx.x >> 5);
  const int mtiles = (M + 15) >> 4;
  const int npairs = (((N + 15) >> 4) + 1) >> 1;   // pairs of 16-wide N tiles
  if (wave >= mtiles * npairs) return;
  const int tm = wave % mtiles, tp = wave / mtiles;
  const int r = lane & 15;   // A m-row / B n-row within tile
  const int h = lane >> 4;   // half-wave
  const __bf16* arow  = A + (size_t)(tm * 16 + r) * lda;
  const __bf16* brow0 = B + (size_t)(tp * 32 + r) * ldb;
  const __bf16* brow1 = brow0 + (size_t)16 * ldb;

  v8f c0 = {}, c1 = {};
  for (int kk = 0; kk < K; kk += 32) {
    // A lane layout: one contiguous 16-element K run [kk+h*16, +16)
    v8bf a0 = *(const v8bf*)(arow + kk + h * 16);
    v8bf a1 = *(const v8bf*)(arow + kk + h * 16 + 8);
    v16bf a = __builtin_shufflevector(a0, a1, 0, 1, 2, 3, 4, 5, 6, 7,
                                      8, 9, 10, 11, 12, 13, 14, 15);
    // B lane layout: one contiguous 16-element K run [kk+h*16, +16)
    v16bf b0 = *(const v16bf*)(brow0 + kk + h * 16);
    v16bf b1 = *(const v16bf*)(brow1 + kk + h * 16);
    c0 = __builtin_amdgcn_wmma_f32_16x16x32_bf16(false, a, false, b0, (short)0, c0,
                                                 false, false);
    c1 = __builtin_amdgcn_wmma_f32_16x16x32_bf16(false, a, false, b1, (short)0, c1,
                                                 false, false);
  }
  // C/D: lane holds n=r, rows m = h*8+vr
  const int nc0 = tp * 32 + r;
  const int nc1 = nc0 + 16;
  if (nc0 < N) {
    float bv = bias ? bias[nc0] : 0.f;
#pragma unroll
    for (int vr = 0; vr < 8; ++vr) {
      int m = tm * 16 + h * 8 + vr;
      if (m < M) Out[(size_t)m * ldo + ocol + nc0] = c0[vr] + bv;
    }
  }
  if (nc1 < N) {
    float bv = bias ? bias[nc1] : 0.f;
#pragma unroll
    for (int vr = 0; vr < 8; ++vr) {
      int m = tm * 16 + h * 8 + vr;
      if (m < M) Out[(size_t)m * ldo + ocol + nc1] = c1[vr] + bv;
    }
  }
}

// ---------------------------------------------------------------------------
// 3) Sequential GRU scan (block 0 = fwd, block 1 = bwd); h in LDS.
// ---------------------------------------------------------------------------
__global__ void gru_scan2(const float* __restrict__ xw_f, const float* __restrict__ xw_b,
                          const float* __restrict__ whh_f, const float* __restrict__ whh_b,
                          const float* __restrict__ bhh_f, const float* __restrict__ bhh_b,
                          float* __restrict__ out, int T) {
  __shared__ float h_s[GRU_H];
  __shared__ float g_s[3 * GRU_H];
  const int dir = blockIdx.x;
  const float* xw  = dir ? xw_b  : xw_f;
  const float* whh = dir ? whh_b : whh_f;
  const float* bhh = dir ? bhh_b : bhh_f;
  const int col_off = dir ? GRU_H : 0;
  const int tid = threadIdx.x;

  if (tid < GRU_H) h_s[tid] = 0.f;
  __syncthreads();

  for (int s = 0; s < T; ++s) {
    const int t = dir ? (T - 1 - s) : s;
    if (tid < 3 * GRU_H) {
      const float* wr = whh + (size_t)tid * GRU_H;
      float g = bhh[tid];
#pragma unroll 4
      for (int c = 0; c < GRU_H; ++c) g += wr[c] * h_s[c];
      g_s[tid] = g;
    }
    __syncthreads();
    if (tid < GRU_H) {
      const float* xwt = xw + (size_t)t * (3 * GRU_H);
      float r  = 1.f / (1.f + __expf(-(xwt[tid] + g_s[tid])));
      float z  = 1.f / (1.f + __expf(-(xwt[GRU_H + tid] + g_s[GRU_H + tid])));
      float cg = fast_tanh(xwt[2 * GRU_H + tid] + r * g_s[2 * GRU_H + tid]);
      float hn = (1.f - z) * cg + z * h_s[tid];
      out[(size_t)t * (2 * GRU_H) + col_off + tid] = hn;
      h_s[tid] = hn;
    }
    __syncthreads();
  }
}

// ---------------------------------------------------------------------------
// 4) Fused pairwise scorer (guard-free, all operands pre-padded):
//    score(i,j) = W3 . relu(W2 @ tanh(s1[i]+s2[j]+b1) + b2) + b3, diag zeroed.
//    One wave = 1 i x 16 j; tanh fused into A-tile build (hp never in memory);
//    two independent WMMA chains (N tiles 0-15 / 16-31).
// ---------------------------------------------------------------------------
__global__ void pairwise_scores(const float* __restrict__ s1, const float* __restrict__ s2,
                                const float* __restrict__ b1p, const __bf16* __restrict__ W2p,
                                const float* __restrict__ b2p, const float* __restrict__ W3p,
                                const float* __restrict__ b3, float* __restrict__ scores,
                                int n) {
  const int lane = threadIdx.x & 31;
  const int wv   = threadIdx.x >> 5;
  const int i  = blockIdx.y * 8 + wv;
  const int j0 = blockIdx.x * 16;
  const int r  = lane & 15;
  const int h  = lane >> 4;
  const float* s1r = s1 + (size_t)i * SPAD;
  const float* s2r = s2 + (size_t)(j0 + r) * SPAD;
  const int n0 = r, n1 = 16 + r;
  const __bf16* w2a = W2p + (size_t)n0 * SPAD;
  const __bf16* w2b = W2p + (size_t)n1 * SPAD;

  v8f c0 = {}, c1 = {};
#pragma unroll
  for (int kk = 0; kk < SPAD; kk += 32) {
    v16bf a;
#pragma unroll
    for (int half = 0; half < 2; ++half) {
      const int base = kk + h * 16 + half * 8;
      v4f u0 = *(const v4f*)(s1r + base);
      v4f u1 = *(const v4f*)(s1r + base + 4);
      v4f w0 = *(const v4f*)(s2r + base);
      v4f w1 = *(const v4f*)(s2r + base + 4);
      v4f z0 = *(const v4f*)(b1p + base);
      v4f z1 = *(const v4f*)(b1p + base + 4);
#pragma unroll
      for (int e = 0; e < 4; ++e) {
        a[half * 8 + e]     = (__bf16)fast_tanh(u0[e] + w0[e] + z0[e]);
        a[half * 8 + 4 + e] = (__bf16)fast_tanh(u1[e] + w1[e] + z1[e]);
      }
    }
    v16bf bA = *(const v16bf*)(w2a + kk + h * 16);
    v16bf bB = *(const v16bf*)(w2b + kk + h * 16);
    c0 = __builtin_amdgcn_wmma_f32_16x16x32_bf16(false, a, false, bA, (short)0, c0,
                                                 false, false);
    c1 = __builtin_amdgcn_wmma_f32_16x16x32_bf16(false, a, false, bB, (short)0, c1,
                                                 false, false);
  }

  // Epilogue: relu + W3 dot; reduce over the 16 n-lanes of each half-wave.
  float w3a = W3p[n0], b2a = b2p[n0];
  float w3b = W3p[n1], b2b = b2p[n1];
  float acc[8];
#pragma unroll
  for (int vr = 0; vr < 8; ++vr) {
    float g0 = c0[vr] + b2a; g0 = g0 > 0.f ? g0 : 0.f;
    float g1 = c1[vr] + b2b; g1 = g1 > 0.f ? g1 : 0.f;
    acc[vr] = w3a * g0 + w3b * g1;
  }
#pragma unroll
  for (int off = 1; off < 16; off <<= 1) {
#pragma unroll
    for (int vr = 0; vr < 8; ++vr) acc[vr] += __shfl_xor(acc[vr], off, 32);
  }
  if (r == 0) {
    float b3v = b3[0];
#pragma unroll
    for (int vr = 0; vr < 8; ++vr) {
      int jj = j0 + h * 8 + vr;
      float sc = acc[vr] + b3v;
      if (i == jj) sc = 0.f;
      scores[(size_t)i * n + jj] = sc;
    }
  }
}

// ---------------------------------------------------------------------------
// 5) Softmax over axis 0 (per column). One wave per column.
// ---------------------------------------------------------------------------
__global__ void softmax_axis0(const float* __restrict__ s, float* __restrict__ out, int n) {
  const int lane = threadIdx.x & 31;
  const int col  = blockIdx.x * (blockDim.x >> 5) + (threadIdx.x >> 5);
  if (col >= n) return;
  float mx = -1e30f;
  for (int i = lane; i < n; i += 32) mx = fmaxf(mx, s[(size_t)i * n + col]);
#pragma unroll
  for (int off = 16; off; off >>= 1) mx = fmaxf(mx, __shfl_xor(mx, off, 32));
  float sum = 0.f;
  for (int i = lane; i < n; i += 32) sum += __expf(s[(size_t)i * n + col] - mx);
#pragma unroll
  for (int off = 16; off; off >>= 1) sum += __shfl_xor(sum, off, 32);
  float inv = 1.f / sum;
  for (int i = lane; i < n; i += 32)
    out[(size_t)i * n + col] = __expf(s[(size_t)i * n + col] - mx) * inv;
}

// ---------------------------------------------------------------------------
// Host driver
// ---------------------------------------------------------------------------
static inline void launch_pack(const float* src, int lds, int col_off, int rows, int cols,
                               __bf16* dst, int ldd, int drows, hipStream_t stream) {
  int total = drows * ldd;
  pack_bf16<<<(total + 255) / 256, 256, 0, stream>>>(src, lds, col_off, rows, cols,
                                                     dst, ldd, drows);
}

static inline void launch_gemm(const __bf16* A, int lda, const __bf16* B, int ldb,
                               const float* bias, float* Out, int ldo, int ocol,
                               int M, int N, int K, hipStream_t stream) {
  int mtiles = (M + 15) / 16;
  int npairs = ((N + 15) / 16 + 1) / 2;
  int waves = mtiles * npairs;
  int blocks = (waves + 7) / 8;   // 8 waves per 256-thread block
  wmma_gemm_bf16<<<blocks, 256, 0, stream>>>(A, lda, B, ldb, bias, Out, ldo, ocol, M, N, K);
}

extern "C" void kernel_launch(void* const* d_in, const int* in_sizes, int n_in,
                              void* d_out, int out_size, void* d_ws, size_t ws_size,
                              hipStream_t stream) {
  const int T = SEQ_T;
  const int* tok = (const int*)d_in[0];
  const int* pos = (const int*)d_in[1];
  const float* wemb = (const float*)d_in[2];
  const float* pemb = (const float*)d_in[3];
  const float* w_ih_l0  = (const float*)d_in[4];
  const float* w_hh_l0  = (const float*)d_in[5];
  const float* b_ih_l0  = (const float*)d_in[6];
  const float* b_hh_l0  = (const float*)d_in[7];
  const float* w_ih_l0r = (const float*)d_in[8];
  const float* w_hh_l0r = (const float*)d_in[9];
  const float* b_ih_l0r = (const float*)d_in[10];
  const float* b_hh_l0r = (const float*)d_in[11];
  const float* w_ih_l1  = (const float*)d_in[12];
  const float* w_hh_l1  = (const float*)d_in[13];
  const float* b_ih_l1  = (const float*)d_in[14];
  const float* b_hh_l1  = (const float*)d_in[15];
  const float* w_ih_l1r = (const float*)d_in[16];
  const float* w_hh_l1r = (const float*)d_in[17];
  const float* b_ih_l1r = (const float*)d_in[18];
  const float* b_hh_l1r = (const float*)d_in[19];
  const float* W1 = (const float*)d_in[20];
  const float* b1 = (const float*)d_in[21];
  const float* W2 = (const float*)d_in[22];
  const float* b2 = (const float*)d_in[23];
  const float* W3 = (const float*)d_in[24];
  const float* b3 = (const float*)d_in[25];

  // ---- workspace layout (f32 region, then bf16 region) ----
  float* ws  = (float*)d_ws;
  float* x   = ws;                   // 512*320
  float* xwf = x   + T * 320;        // 512*600
  float* xwb = xwf + T * 600;        // 512*600
  float* h1  = xwb + T * 600;        // 512*400
  float* h2  = h1  + T * 400;        // 512*400
  float* s1b = h2  + T * 400;        // 512*SPAD (zero-padded)
  float* s2b = s1b + T * SPAD;       // 512*SPAD
  float* raw = s2b + T * SPAD;       // 512*512
  float* b1p = raw + T * T;          // SPAD
  float* b2p = b1p + SPAD;           // 32
  float* W3p = b2p + 32;             // 32
  __bf16* bfbase = (__bf16*)(W3p + 32);
  __bf16* Xp  = bfbase;              // up to 512*416
  __bf16* Wp  = Xp + 512 * 416;      // up to 608*416 (rows padded to mult 32)
  __bf16* W2p = Wp + 608 * 416;      // 32*SPAD

  // 1) embeddings
  gather_x_kernel<<<(T * 320 + 255) / 256, 256, 0, stream>>>(tok, pos, wemb, pemb, x, T);

  // 2) layer 0: pack + GEMMs (K=320 already /32) + scans
  launch_pack(x, 320, 0, T, 320, Xp, 320, T, stream);
  launch_pack(w_ih_l0,  320, 0, 600, 320, Wp, 320, 608, stream);
  launch_gemm(Xp, 320, Wp, 320, b_ih_l0,  xwf, 600, 0, T, 600, 320, stream);
  launch_pack(w_ih_l0r, 320, 0, 600, 320, Wp, 320, 608, stream);
  launch_gemm(Xp, 320, Wp, 320, b_ih_l0r, xwb, 600, 0, T, 600, 320, stream);
  gru_scan2<<<2, 640, 0, stream>>>(xwf, xwb, w_hh_l0, w_hh_l0r, b_hh_l0, b_hh_l0r, h1, T);

  // 3) layer 1: pack (400 -> 416) + GEMMs + scans
  launch_pack(h1, 400, 0, T, 400, Xp, 416, T, stream);
  launch_pack(w_ih_l1,  400, 0, 600, 400, Wp, 416, 608, stream);
  launch_gemm(Xp, 416, Wp, 416, b_ih_l1,  xwf, 600, 0, T, 600, 416, stream);
  launch_pack(w_ih_l1r, 400, 0, 600, 400, Wp, 416, 608, stream);
  launch_gemm(Xp, 416, Wp, 416, b_ih_l1r, xwb, 600, 0, T, 600, 416, stream);
  gru_scan2<<<2, 640, 0, stream>>>(xwf, xwb, w_hh_l1, w_hh_l1r, b_hh_l1, b_hh_l1r, h2, T);

  // 4) biaffine projections into zero-padded s1/s2 (ld = SPAD)
  //    B rows padded to 224 (mult of 32) for the paired-N-tile GEMM.
  fill_zero_f32<<<(2 * T * SPAD + 255) / 256, 256, 0, stream>>>(s1b, 2 * T * SPAD);
  launch_pack(h2, 400, 0, T, 400, Xp, 416, T, stream);
  launch_pack(W1, 800, 0,   200, 400, Wp, 416, 224, stream);
  launch_gemm(Xp, 416, Wp, 416, nullptr, s1b, SPAD, 0, T, 200, 416, stream);
  launch_pack(W1, 800, 400, 200, 400, Wp, 416, 224, stream);
  launch_gemm(Xp, 416, Wp, 416, nullptr, s2b, SPAD, 0, T, 200, 416, stream);

  // 5) pad small operands for the guard-free pairwise kernel
  pack_pad_f32<<<1, 256, 0, stream>>>(b1, 200, b1p, SPAD);
  pack_pad_f32<<<1, 64, 0, stream>>>(b2, 20, b2p, 32);
  pack_pad_f32<<<1, 64, 0, stream>>>(W3, 20, W3p, 32);
  launch_pack(W2, 200, 0, 20, 200, W2p, SPAD, 32, stream);

  // 6) fused pairwise tanh -> WMMA -> relu -> W3 dot
  {
    dim3 grid(T / 16, T / 8);
    pairwise_scores<<<grid, 256, 0, stream>>>(s1b, s2b, b1p, W2p, b2p, W3p, b3, raw, T);
  }

  // 7) softmax over axis 0
  softmax_axis0<<<(T + 7) / 8, 256, 0, stream>>>(raw, (float*)d_out, T);
}
